// TrainingMLPModel_47631187313215
// MI455X (gfx1250) — compile-verified, matrix operand order checked by reading, not run on hardware
//
#include <hip/hip_runtime.h>
#include <hip/hip_bf16.h>

// ---------------------------------------------------------------------------
// Problem dims (from reference): B,N,D,H,O,A,L
// ---------------------------------------------------------------------------
constexpr int kB = 128, kN = 100, kD = 2048, kH = 1024;
constexpr int kO = 1600, kA = 400, kL = 2048;
constexpr int kM = kB * kN;            // 12800 flattened rows

typedef __bf16 bf16_t;
typedef __attribute__((ext_vector_type(16))) __bf16 v16bf;
typedef __attribute__((ext_vector_type(8)))  __bf16 v8bf;
typedef __attribute__((ext_vector_type(8)))  float  v8f;

__device__ __forceinline__ bf16_t f2bf(float f) {
  unsigned int u = __builtin_bit_cast(unsigned int, f);
  u += 0x7FFFu + ((u >> 16) & 1u);                 // round-to-nearest-even
  unsigned short h = (unsigned short)(u >> 16);
  return __builtin_bit_cast(bf16_t, h);
}

__device__ __forceinline__ v16bf cat8(v8bf lo, v8bf hi) {
  return __builtin_shufflevector(lo, hi, 0,1,2,3,4,5,6,7,8,9,10,11,12,13,14,15);
}

// ---------------------------------------------------------------------------
// f32 -> bf16 elementwise convert (for x)
// ---------------------------------------------------------------------------
__global__ void convert_f32_bf16(const float* __restrict__ in,
                                 bf16_t* __restrict__ out, long n) {
  long i = (long)blockIdx.x * blockDim.x + threadIdx.x;
  long stride = (long)gridDim.x * blockDim.x;
  for (; i < n; i += stride) out[i] = f2bf(in[i]);
}

// ---------------------------------------------------------------------------
// Transpose + convert: in (R x C) f32 row-major -> out (C x R) bf16 row-major.
// 32x32 LDS tile, padded to kill bank conflicts.
// ---------------------------------------------------------------------------
__global__ __launch_bounds__(256)
void transpose_convert(const float* __restrict__ in, bf16_t* __restrict__ out,
                       int R, int C) {
  __shared__ float tile[32][33];
  const int c0 = blockIdx.x * 32;
  const int r0 = blockIdx.y * 32;
  const int tx = threadIdx.x;     // 0..31
  const int ty = threadIdx.y;     // 0..7
  for (int i = ty; i < 32; i += 8) {
    int r = r0 + i, c = c0 + tx;
    tile[i][tx] = (r < R && c < C) ? in[(size_t)r * C + c] : 0.0f;
  }
  __syncthreads();
  for (int i = ty; i < 32; i += 8) {
    int c = c0 + i, r = r0 + tx;
    if (c < C && r < R) out[(size_t)c * R + r] = f2bf(tile[tx][i]);
  }
}

// ---------------------------------------------------------------------------
// WMMA GEMM:  C(M x N) = A(M x K) * Bt(N x K)^T  (+bias) [+relu]
// A, Bt bf16 row-major; each wave computes a 32x64 tile with 8 accumulators.
// Fragment layout per CDNA5 ISA 16-bit A/B tables:
//   lane L: m (or n) = L & 15; element e <-> K = (e&7) + ((e>>3)<<4) + 8*(L>=16)
// so each lane loads two contiguous 16-byte chunks per fragment.
//
// Two-stage software pipeline with disjoint ping/pong fragment registers:
// loads for chunk k+32 never overwrite registers still being read by the
// in-flight WMMAs of chunk k (avoids WMMA WAR-hazard v_nop padding,
// ISA 7.12.1) and lets the scheduler keep loads in flight across the MMAs.
// ---------------------------------------------------------------------------
__device__ __forceinline__ void load_frags(const bf16_t* __restrict__ a0,
                                           const bf16_t* __restrict__ a1,
                                           const bf16_t* const brow[4], int k0,
                                           v16bf af[2], v16bf bfr[4]) {
  af[0] = cat8(*(const v8bf*)(a0 + k0), *(const v8bf*)(a0 + k0 + 16));
  af[1] = cat8(*(const v8bf*)(a1 + k0), *(const v8bf*)(a1 + k0 + 16));
#pragma unroll
  for (int j = 0; j < 4; ++j)
    bfr[j] = cat8(*(const v8bf*)(brow[j] + k0), *(const v8bf*)(brow[j] + k0 + 16));
}

__device__ __forceinline__ void do_mma(const v16bf af[2], const v16bf bfr[4],
                                       v8f acc[2][4]) {
#pragma unroll
  for (int j = 0; j < 4; ++j) {
    acc[0][j] = __builtin_amdgcn_wmma_f32_16x16x32_bf16(
        false, af[0], false, bfr[j], (short)0, acc[0][j], false, false);
    acc[1][j] = __builtin_amdgcn_wmma_f32_16x16x32_bf16(
        false, af[1], false, bfr[j], (short)0, acc[1][j], false, false);
  }
}

template<bool RELU, bool STOREBF>
__global__ __launch_bounds__(128)
void wmma_gemm(const bf16_t* __restrict__ Abuf, const bf16_t* __restrict__ Bt,
               const float* __restrict__ bias, void* __restrict__ Cout,
               int M, int N, int K) {
  const int lane = threadIdx.x & 31;
  const int wave = threadIdx.x >> 5;
  const int hl   = lane >> 4;        // half-wave: 0 or 1
  const int l16  = lane & 15;
  const int mBase = blockIdx.x * 128 + wave * 32;
  const int nBase = blockIdx.y * 64;
  if (mBase >= M) return;

  v8f acc[2][4];
#pragma unroll
  for (int i = 0; i < 2; ++i)
#pragma unroll
    for (int j = 0; j < 4; ++j)
#pragma unroll
      for (int r = 0; r < 8; ++r) acc[i][j][r] = 0.0f;

  const bf16_t* arow0 = Abuf + (size_t)(mBase + l16) * K + (hl << 3);
  const bf16_t* arow1 = arow0 + (size_t)16 * K;

  // N is always a multiple of 16: each 16-wide B fragment is entirely valid or
  // entirely invalid. Invalid fragments use a clamped in-bounds pointer and
  // their accumulators are never stored.
  const bf16_t* brow[4];
#pragma unroll
  for (int j = 0; j < 4; ++j) {
    int col = nBase + j * 16 + l16;
    brow[j] = Bt + (size_t)(col < N ? col : l16) * K + (hl << 3);
  }

  v16bf afA[2], bfA[4], afB[2], bfB[4];
  load_frags(arow0, arow1, brow, 0, afA, bfA);

  for (int k0 = 0; k0 < K; k0 += 64) {          // K % 64 == 0 for all layers
    __builtin_prefetch(arow0 + k0 + 128, 0, 1);
    __builtin_prefetch(arow1 + k0 + 128, 0, 1);

    load_frags(arow0, arow1, brow, k0 + 32, afB, bfB);
    do_mma(afA, bfA, acc);

    const int kn = (k0 + 64 < K) ? k0 + 64 : 0; // clamp: redundant but in-bounds
    load_frags(arow0, arow1, brow, kn, afA, bfA);
    do_mma(afB, bfB, acc);
  }

  // Epilogue: C/D layout -> VGPR r holds (row = r + 8*hl, col = l16) of 16x16 tile.
#pragma unroll
  for (int j = 0; j < 4; ++j) {
    const int colBase = nBase + j * 16;         // scalar (uniform) validity test
    if (colBase < N) {
      const int col = colBase + l16;
      const float bv = bias ? bias[col] : 0.0f;
#pragma unroll
      for (int i = 0; i < 2; ++i) {
#pragma unroll
        for (int r = 0; r < 8; ++r) {
          int row = mBase + i * 16 + (hl << 3) + r;
          float v = acc[i][j][r] + bv;
          if (RELU) v = v > 0.0f ? v : 0.0f;
          if (STOREBF) ((bf16_t*)Cout)[(size_t)row * N + col] = f2bf(v);
          else         ((float*)Cout)[(size_t)row * N + col]  = v;
        }
      }
    }
  }
}

// ---------------------------------------------------------------------------
// Row softmax in place: one block per row.
// ---------------------------------------------------------------------------
__global__ __launch_bounds__(256)
void softmax_rows(float* __restrict__ data, int cols) {
  __shared__ float red[256];
  const int row = blockIdx.x;
  const int t = threadIdx.x;
  float* p = data + (size_t)row * cols;

  float m = -3.4e38f;
  for (int c = t; c < cols; c += 256) m = fmaxf(m, p[c]);
  red[t] = m; __syncthreads();
  for (int s = 128; s > 0; s >>= 1) { if (t < s) red[t] = fmaxf(red[t], red[t + s]); __syncthreads(); }
  m = red[0]; __syncthreads();

  float sum = 0.0f;
  for (int c = t; c < cols; c += 256) { float e = __expf(p[c] - m); p[c] = e; sum += e; }
  red[t] = sum; __syncthreads();
  for (int s = 128; s > 0; s >>= 1) { if (t < s) red[t] += red[t + s]; __syncthreads(); }
  float inv = 1.0f / red[0];
  for (int c = t; c < cols; c += 256) p[c] *= inv;
}

// ---------------------------------------------------------------------------
// Per-label argmax over the 100 descriptors + gather of both distributions.
// One block per label l (L = 2048). Matches jnp.argmax first-index tie-break.
// ---------------------------------------------------------------------------
__global__ __launch_bounds__(256)
void select_gather(const float* __restrict__ po, const float* __restrict__ pa,
                   const int* __restrict__ label_img,
                   const int* __restrict__ obj_labels,
                   const int* __restrict__ att_labels,
                   const int* __restrict__ num_descs,
                   float* __restrict__ obj_out, float* __restrict__ att_out) {
  __shared__ float ss[256];
  __shared__ int   si[256];
  const int l = blockIdx.x;
  const int t = threadIdx.x;
  const int b  = label_img[l];
  const int nd = num_descs[b];
  const int ol = obj_labels[l];
  const int al = att_labels[l];
  const float NEG = -__builtin_huge_valf();

  float best = NEG; int bi = 0x7fffffff;
  for (int n = t; n < kN; n += 256) {
    float s = NEG;
    if (n < nd) {
      float a = po[((size_t)b * kN + n) * kO + ol];
      float c = pa[((size_t)b * kN + n) * kA + al];
      s = a * c;
    }
    if (s > best || (s == best && n < bi)) { best = s; bi = n; }
  }
  ss[t] = best; si[t] = bi; __syncthreads();
  for (int s = 128; s > 0; s >>= 1) {
    if (t < s) {
      float os = ss[t + s]; int oi = si[t + s];
      if (os > ss[t] || (os == ss[t] && oi < si[t])) { ss[t] = os; si[t] = oi; }
    }
    __syncthreads();
  }
  const int desc = (si[0] == 0x7fffffff) ? 0 : si[0];

  const float* srco = po + ((size_t)b * kN + desc) * kO;
  for (int c = t; c < kO; c += 256) obj_out[(size_t)l * kO + c] = srco[c];
  const float* srca = pa + ((size_t)b * kN + desc) * kA;
  for (int c = t; c < kA; c += 256) att_out[(size_t)l * kA + c] = srca[c];
}

// ---------------------------------------------------------------------------
// Host-side orchestration
// ---------------------------------------------------------------------------
extern "C" void kernel_launch(void* const* d_in, const int* in_sizes, int n_in,
                              void* d_out, int out_size, void* d_ws, size_t ws_size,
                              hipStream_t stream) {
  (void)in_sizes; (void)out_size; (void)ws_size;

  const float* x    = (const float*)d_in[0];
  const int* num_descs = (const int*)d_in[1];
  const int* label_img = (const int*)d_in[2];
  const int* obj_labels = (const int*)d_in[3];
  const int* att_labels = (const int*)d_in[4];
  const float* w1o = (const float*)d_in[5];
  const float* b1o = (const float*)d_in[6];
  const float* w2o = (const float*)d_in[7];
  const float* b2o = (const float*)d_in[8];
  const float* w3o = (const float*)d_in[9];
  const float* b3o = (const float*)d_in[10];
  const float* w1a = (const float*)d_in[11];
  const float* b1a = (const float*)d_in[12];
  const float* w2a = (const float*)d_in[13];
  const float* b2a = (const float*)d_in[14];
  const float* w3a = (const float*)d_in[15];
  const float* b3a = (n_in > 16) ? (const float*)d_in[16] : nullptr; // zeros in setup

  // Workspace layout (256B aligned); total ~224 MB.
  char* ws = (char*)d_ws;
  size_t off = 0;
  auto take = [&](size_t bytes) -> char* {
    char* p = ws + off;
    off += (bytes + 255) & ~(size_t)255;
    return p;
  };
  bf16_t* xb   = (bf16_t*)take((size_t)kM * kD * 2);
  bf16_t* w1oT = (bf16_t*)take((size_t)kH * kD * 2);
  bf16_t* w2oT = (bf16_t*)take((size_t)kH * kH * 2);
  bf16_t* w3oT = (bf16_t*)take((size_t)kO * kH * 2);
  bf16_t* w1aT = (bf16_t*)take((size_t)kH * kD * 2);
  bf16_t* w2aT = (bf16_t*)take((size_t)kH * kH * 2);
  bf16_t* w3aT = (bf16_t*)take((size_t)kA * kH * 2);
  bf16_t* h1   = (bf16_t*)take((size_t)kM * kH * 2);
  bf16_t* h2   = (bf16_t*)take((size_t)kM * kH * 2);
  float*  po   = (float*) take((size_t)kM * kO * 4);
  float*  pa   = (float*) take((size_t)kM * kA * 4);

  // --- converts ---
  {
    long n = (long)kM * kD;
    convert_f32_bf16<<<dim3(4096), dim3(256), 0, stream>>>(x, xb, n);
  }
  dim3 tb(32, 8);
  auto tgrid = [](int R, int C) { return dim3((C + 31) / 32, (R + 31) / 32); };
  transpose_convert<<<tgrid(kD, kH), tb, 0, stream>>>(w1o, w1oT, kD, kH);
  transpose_convert<<<tgrid(kH, kH), tb, 0, stream>>>(w2o, w2oT, kH, kH);
  transpose_convert<<<tgrid(kH, kO), tb, 0, stream>>>(w3o, w3oT, kH, kO);
  transpose_convert<<<tgrid(kD, kH), tb, 0, stream>>>(w1a, w1aT, kD, kH);
  transpose_convert<<<tgrid(kH, kH), tb, 0, stream>>>(w2a, w2aT, kH, kH);
  transpose_convert<<<tgrid(kH, kA), tb, 0, stream>>>(w3a, w3aT, kH, kA);

  // --- GEMM pipeline ---
  const dim3 gblk(128);
  auto ggrid = [](int M, int N) { return dim3(M / 128, (N + 63) / 64); };

  // obj MLP
  wmma_gemm<true,  true ><<<ggrid(kM, kH), gblk, 0, stream>>>(xb, w1oT, b1o, h1, kM, kH, kD);
  wmma_gemm<true,  true ><<<ggrid(kM, kH), gblk, 0, stream>>>(h1, w2oT, b2o, h2, kM, kH, kH);
  wmma_gemm<false, false><<<ggrid(kM, kO), gblk, 0, stream>>>(h2, w3oT, b3o, po, kM, kO, kH);
  softmax_rows<<<dim3(kM), dim3(256), 0, stream>>>(po, kO);

  // att MLP (reuses h1/h2)
  wmma_gemm<true,  true ><<<ggrid(kM, kH), gblk, 0, stream>>>(xb, w1aT, b1a, h1, kM, kH, kD);
  wmma_gemm<true,  true ><<<ggrid(kM, kH), gblk, 0, stream>>>(h1, w2aT, b2a, h2, kM, kH, kH);
  wmma_gemm<false, false><<<ggrid(kM, kA), gblk, 0, stream>>>(h2, w3aT, b3a, pa, kM, kA, kH);
  softmax_rows<<<dim3(kM), dim3(256), 0, stream>>>(pa, kA);

  // --- select + gather ---
  float* obj_out = (float*)d_out;
  float* att_out = obj_out + (size_t)kL * kO;
  select_gather<<<dim3(kL), dim3(256), 0, stream>>>(po, pa, label_img, obj_labels,
                                                    att_labels, num_descs,
                                                    obj_out, att_out);
}
